// DenseAqt_27066883899832
// MI455X (gfx1250) — 8-bit matrix operand order repaired and checked by reading, compiled, not run on hardware
//
#include <hip/hip_runtime.h>

typedef __attribute__((ext_vector_type(8))) int   v8i;
typedef __attribute__((ext_vector_type(4))) float v4f;

#define NN 65536
#define KK 512
#define FF 512
#define KB (KK / 64)
#define FT (FF / 16)
#define A_SCALE (127.0f / 3.0f)
#define RND_MAGIC 12582912.0f

#define WSB_OFF 4096

__global__ __launch_bounds__(256) void wscale_kernel(const float* __restrict__ w,
                                                     float* __restrict__ wscale,
                                                     float* __restrict__ recip) {
    int f = blockIdx.x * 256 + threadIdx.x;
    if (f >= FF) return;
    float bound = 0.0f;
    for (int k = 0; k < KK; ++k) bound = fmaxf(bound, fabsf(w[k * FF + f]));
    bound = fmaxf(bound, 1e-7f);
    float s = 127.0f / bound;
    wscale[f] = s;
    recip[f]  = 1.0f / (A_SCALE * s);
}

__global__ __launch_bounds__(256) void packb_kernel(const float* __restrict__ w,
                                                    const float* __restrict__ wscale,
                                                    int* __restrict__ bq) {
    int t    = blockIdx.x * 256 + threadIdx.x;
    int lane = t & 31;
    int frag = t >> 5;
    int kb    = frag & (KB - 1);
    int ftile = frag / KB;
    int g = lane >> 4;
    int n = ftile * 16 + (lane & 15);
    float s = wscale[n];
#pragma unroll
    for (int wd = 0; wd < 8; ++wd) {
        int kbase = kb * 64 + (wd >> 2) * 32 + g * 16 + (wd & 3) * 4;
        unsigned u[4];
#pragma unroll
        for (int b = 0; b < 4; ++b) {
            float c = fminf(fmaxf(w[(size_t)(kbase + b) * FF + n] * s, -127.0f), 127.0f);
            u[b] = __float_as_uint(c + RND_MAGIC);
        }
        unsigned lo = __builtin_amdgcn_perm(u[1], u[0], 0x00000400u);
        unsigned hi = __builtin_amdgcn_perm(u[3], u[2], 0x00000400u);
        bq[frag * 256 + lane * 8 + wd] =
            (int)__builtin_amdgcn_perm(hi, lo, 0x05040100u);
    }
}

__global__ __launch_bounds__(256) void gemm_kernel(const float* __restrict__ x,
                                                   const int*   __restrict__ bq,
                                                   const float* __restrict__ recip,
                                                   const float* __restrict__ bias,
                                                   float* __restrict__ out) {
    const int tid   = threadIdx.x;
    const int lane  = tid & 31;
    const int wv    = tid >> 5;
    const int msub  = wv >> 2;
    const int fgrp  = wv & 3;
    const int lhalf = lane & 15;
    const int g     = lane >> 4;

    const int    mtile = blockIdx.x * 2 + msub;
    const int    m0    = mtile * 16;
    const size_t arow  = (size_t)(m0 + lhalf) * KK;

    v8i acc[8] = {};

#pragma unroll
    for (int kb = 0; kb < KB; ++kb) {
        v8i a;
#pragma unroll
        for (int wd = 0; wd < 8; ++wd) {
            int koff = kb * 64 + (wd >> 2) * 32 + g * 16 +
                       (wd & 3) * 4;
            v4f xv = __builtin_nontemporal_load((const v4f*)(x + arow + koff));
            unsigned u[4];
#pragma unroll
            for (int b = 0; b < 4; ++b) {
                float c = fminf(fmaxf(xv[b], -3.0f), 3.0f);
                u[b] = __float_as_uint(fmaf(c, A_SCALE, RND_MAGIC));
            }
            unsigned lo = __builtin_amdgcn_perm(u[1], u[0], 0x00000400u);
            unsigned hi = __builtin_amdgcn_perm(u[3], u[2], 0x00000400u);
            a[wd] = (int)__builtin_amdgcn_perm(hi, lo, 0x05040100u);
        }
#pragma unroll
        for (int ft = 0; ft < 8; ++ft) {
            int frag = (fgrp * 8 + ft) * KB + kb;
            v8i b = *(const v8i*)(bq + (size_t)frag * 256 + lane * 8);
            acc[ft] = __builtin_amdgcn_wmma_i32_16x16x64_iu8(
                 true, a,  true, b, acc[ft],
                 false,  false);
        }
    }

#pragma unroll
    for (int ft = 0; ft < 8; ++ft) {
        int   n  = (fgrp * 8 + ft) * 16 + lhalf;
        float s  = recip[n];
        float bb = bias[n];
#pragma unroll
        for (int r = 0; r < 8; ++r) {
            int   row = m0 + r + g * 8;
            float v   = (float)acc[ft][r] * s + bb;
            __builtin_nontemporal_store(v, out + (size_t)row * FF + n);
        }
    }
}

extern "C" void kernel_launch(void* const* d_in, const int* in_sizes, int n_in,
                              void* d_out, int out_size, void* d_ws, size_t ws_size,
                              hipStream_t stream) {
    const float* x      = (const float*)d_in[0];
    const float* kernel = (const float*)d_in[1];
    const float* bias   = (const float*)d_in[2];
    float* out = (float*)d_out;

    float* wscale = (float*)d_ws;
    float* recip  = wscale + FF;
    int*   bq     = (int*)((char*)d_ws + WSB_OFF);

    wscale_kernel<<<(FF + 255) / 256, 256, 0, stream>>>(kernel, wscale, recip);
    packb_kernel<<<(FT * KB * 32) / 256, 256, 0, stream>>>(kernel, wscale, bq);
    gemm_kernel<<<NN / 32, 256, 0, stream>>>(x, bq, recip, bias, out);

    (void)in_sizes; (void)n_in; (void)out_size; (void)ws_size;
}
